// DummyAutoregressiveDecoder_63264868270759
// MI455X (gfx1250) — matrix operand order repaired and checked by reading, not compiled, not run on hardware
//
#include <hip/hip_runtime.h>

// ---------------------------------------------------------------------------
// Sizes from the reference
// ---------------------------------------------------------------------------
#define H      512
#define H3     1536
#define FLAT   65792          // 2*128*257
#define CD     256
#define SD     256
#define D_IN   66304          // CD + SD + FLAT
#define Bsz    4
#define Ssz    16
#define BS     64             // Bsz*Ssz

typedef __bf16 bf16_t;
typedef __attribute__((ext_vector_type(16))) __bf16 v16bf;
typedef __attribute__((ext_vector_type(8)))  __bf16 v8bf;
typedef __attribute__((ext_vector_type(4)))  __bf16 v4bf;
typedef __attribute__((ext_vector_type(8)))  float  v8f;

// WMMA D = A(16x32 bf16) * B(32x16 bf16) + C(16x16 f32)
__device__ __forceinline__ v8f wmma_bf16(v16bf a, v16bf b, v8f c) {
    return __builtin_amdgcn_wmma_f32_16x16x32_bf16(
        /*neg_a=*/false, a, /*neg_b=*/false, b,
        /*c_mod=*/(short)0, c, /*reuse_a=*/false, /*reuse_b=*/false);
}

__device__ __forceinline__ v4bf cvt4(float4 f) {
    return (v4bf){(bf16_t)f.x, (bf16_t)f.y, (bf16_t)f.z, (bf16_t)f.w};
}

// A-fragment loader (16 rows x 32 k, bf16, row-major with given stride).
// Lane L<16 : row=L,    k in {0..15}
// Lane L>=16: row=L-16, k in {16..31}
__device__ __forceinline__ v16bf load_frag_a(const bf16_t* base, int stride, int lane) {
    int r  = lane & 15;
    int k0 = (lane >> 4) << 4;        // 0 or 16
    union { v16bf v; v8bf h[2]; } u;
    u.h[0] = *(const v8bf*)(base + r * stride + k0);
    u.h[1] = *(const v8bf*)(base + r * stride + k0 + 8);
    return u.v;
}

// B-fragment loader (32 k x 16 cols) from memory stored col-major [col][k].
// Lane L<16 : col=L, k=0..15 ; Lane L>=16: col=L-16, k=16..31
__device__ __forceinline__ v16bf load_frag_b(const bf16_t* base, int stride, int lane) {
    int c  = lane & 15;
    int k0 = (lane >> 4) << 4;        // 0 or 16
    union { v16bf v; v8bf h[2]; } u;
    u.h[0] = *(const v8bf*)(base + c * stride + k0);
    u.h[1] = *(const v8bf*)(base + c * stride + k0 + 8);
    return u.v;
}

// ---------------------------------------------------------------------------
// Kernel 0: zero the atomic accumulators (w0, wb)
// ---------------------------------------------------------------------------
__global__ void k_zero(float* p, int n) {
    int i = blockIdx.x * blockDim.x + threadIdx.x;
    if (i < n) p[i] = 0.0f;
}

// ---------------------------------------------------------------------------
// Kernel 1: M[1536][512] = W_ih[:,512:] @ W_proj   (K = 65792)
//           fused: w0 = W_ih[:,512:] @ start_token, wb = .. @ b_proj
// Block: 512 thr (16 waves). WG tile 256(M) x 128(N), wave tile 32x64.
// Grid (6, 4). K staged in LDS as bf16 (32/chunk) with float4 global loads,
// register double-buffering so the next chunk's HBM loads overlap the WMMAs.
// W_ih (407 MB) is streamed from HBM exactly once (~17 us at 23.3 TB/s).
// ---------------------------------------------------------------------------
__global__ __launch_bounds__(512)
void k_gemm_M(const float* __restrict__ Wih, const float* __restrict__ Wproj,
              const float* __restrict__ start_tok, const float* __restrict__ bproj,
              float* __restrict__ Mmat, float* __restrict__ w0, float* __restrict__ wb) {
    __shared__ bf16_t Al[256 * 32];     // [row][k]
    __shared__ bf16_t Bl[128 * 32];     // [col][k]  (transposed stage)
    __shared__ float  stc[32], bpc[32];

    const int tid  = threadIdx.x;
    const int lane = tid & 31;
    const int w    = tid >> 5;
    const int wave_m = (w >> 1) * 32;   // 0..224
    const int wave_n = (w & 1) * 64;    // 0 or 64
    const int m_base = blockIdx.x * 256;
    const int n_base = blockIdx.y * 128;
    const bool do_aux = (blockIdx.y == 0);

    v8f acc[2][4];
#pragma unroll
    for (int mi = 0; mi < 2; ++mi)
#pragma unroll
        for (int ni = 0; ni < 4; ++ni) acc[mi][ni] = (v8f){};

    float aw0 = 0.0f, awb = 0.0f;
    const int rowL = tid >> 1;          // 0..255
    const int half = tid & 1;           // which 16-k half of the chunk

    // ---- register prefetch buffers ----
    float4 aP[4];                       // A: 2048 float4 / 512 thr = 4
    float4 bP[2];                       // B: 1024 float4 / 512 thr = 2
    float  sbP = 0.0f;                  // start_token / b_proj element

    auto prefetch = [&](int kc) {
#pragma unroll
        for (int j = 0; j < 4; ++j) {
            int i = tid + j * 512;       // 0..2047
            int row = i >> 3, k4 = (i & 7) << 2;
            aP[j] = *(const float4*)(Wih + (size_t)(m_base + row) * D_IN +
                                     (CD + SD) + kc + k4);
        }
#pragma unroll
        for (int j = 0; j < 2; ++j) {
            int i = tid + j * 512;       // 0..1023
            int k = i >> 5, n4 = (i & 31) << 2;
            bP[j] = *(const float4*)(Wproj + (size_t)(kc + k) * H + n_base + n4);
        }
        if (tid < 32)      sbP = start_tok[kc + tid];
        else if (tid < 64) sbP = bproj[kc + tid - 32];
    };
    auto stage = [&]() {
#pragma unroll
        for (int j = 0; j < 4; ++j) {
            int i = tid + j * 512;
            int row = i >> 3, k4 = (i & 7) << 2;
            *(v4bf*)(&Al[row * 32 + k4]) = cvt4(aP[j]);
        }
#pragma unroll
        for (int j = 0; j < 2; ++j) {
            int i = tid + j * 512;
            int k = i >> 5, n4 = (i & 31) << 2;
            Bl[(n4 + 0) * 32 + k] = (bf16_t)bP[j].x;
            Bl[(n4 + 1) * 32 + k] = (bf16_t)bP[j].y;
            Bl[(n4 + 2) * 32 + k] = (bf16_t)bP[j].z;
            Bl[(n4 + 3) * 32 + k] = (bf16_t)bP[j].w;
        }
        if (tid < 32)      stc[tid]      = sbP;
        else if (tid < 64) bpc[tid - 32] = sbP;
    };

    prefetch(0);
    for (int kc = 0; kc < FLAT; kc += 32) {
        stage();
        __syncthreads();
        if (kc + 32 < FLAT) prefetch(kc + 32);   // overlaps with WMMAs below

        v16bf af[2], bfr[4];
#pragma unroll
        for (int mi = 0; mi < 2; ++mi)
            af[mi] = load_frag_a(&Al[(wave_m + mi * 16) * 32], 32, lane);
#pragma unroll
        for (int ni = 0; ni < 4; ++ni)
            bfr[ni] = load_frag_b(&Bl[(wave_n + ni * 16) * 32], 32, lane);
#pragma unroll
        for (int mi = 0; mi < 2; ++mi)
#pragma unroll
            for (int ni = 0; ni < 4; ++ni)
                acc[mi][ni] = wmma_bf16(af[mi], bfr[ni], acc[mi][ni]);

        if (do_aux) {
            const bf16_t* ar = &Al[rowL * 32 + half * 16];
#pragma unroll
            for (int j = 0; j < 16; ++j) {
                float a = (float)ar[j];
                aw0 += a * stc[half * 16 + j];
                awb += a * bpc[half * 16 + j];
            }
        }
        __syncthreads();
    }

    // store the M tile (C layout: VGPR r -> row r (+8 for hi half), lane&15 -> col)
#pragma unroll
    for (int mi = 0; mi < 2; ++mi)
#pragma unroll
        for (int ni = 0; ni < 4; ++ni)
#pragma unroll
            for (int r = 0; r < 8; ++r) {
                int row = m_base + wave_m + mi * 16 + r + ((lane >> 4) << 3);
                int col = n_base + wave_n + ni * 16 + (lane & 15);
                Mmat[row * H + col] = acc[mi][ni][r];
            }

    if (do_aux) {
        atomicAdd(&w0[m_base + rowL], aw0);
        atomicAdd(&wb[m_base + rowL], awb);
    }
}

// ---------------------------------------------------------------------------
// Kernel 2: GIcs[bs][g] = b_ih[g] + W_ih[g,0:256]@content[b,s] + W_ih[g,256:512]@style[b]
// grid 64 (one block per (b,s)), 256 threads. The 3 MB W_ih column slab is L2-resident.
// ---------------------------------------------------------------------------
__global__ __launch_bounds__(256)
void k_gi_static(const float* __restrict__ content, const float* __restrict__ style,
                 const float* __restrict__ Wih, const float* __restrict__ bih,
                 float* __restrict__ GIcs) {
    __shared__ float x[CD + SD];
    const int bs  = blockIdx.x;
    const int b   = bs >> 4;
    const int tid = threadIdx.x;
    x[tid]      = content[bs * CD + tid];
    x[CD + tid] = style[b * SD + tid];
    __syncthreads();
    const float4* x4 = (const float4*)x;
    for (int g = tid; g < H3; g += 256) {
        const float4* wr = (const float4*)(Wih + (size_t)g * D_IN);
        float a = bih[g];
#pragma unroll 4
        for (int k = 0; k < (CD + SD) / 4; ++k) {
            float4 wv = wr[k], xv = x4[k];
            a += wv.x * xv.x + wv.y * xv.y + wv.z * xv.z + wv.w * xv.w;
        }
        GIcs[bs * H3 + g] = a;
    }
}

// ---------------------------------------------------------------------------
// Kernel 3: one GRU step (launched 16x; kernel boundary = global sync).
// grid 4 (one block per batch), 512 threads (thread j owns hidden unit j).
// gi = GIcs + (s==0 ? w0 : wb + M@h_prev),  gh = b_hh + W_hh@h_prev
// ---------------------------------------------------------------------------
__global__ __launch_bounds__(512)
void k_gru_step(const float* __restrict__ Mmat, const float* __restrict__ Whh,
                const float* __restrict__ bhh,  const float* __restrict__ GIcs,
                const float* __restrict__ w0,   const float* __restrict__ wb,
                float* __restrict__ Hs, int s) {
    __shared__ float hp[H];
    const int b  = blockIdx.x;
    const int j  = threadIdx.x;
    const int bs = b * Ssz + s;
    hp[j] = (s == 0) ? 0.0f : Hs[(bs - 1) * H + j];
    __syncthreads();
    const float4* hp4 = (const float4*)hp;

    float gi[3], gh[3];
#pragma unroll
    for (int q = 0; q < 3; ++q) {
        const int row = q * H + j;
        float tg = GIcs[bs * H3 + row];
        float th = bhh[row];
        const float4* whr = (const float4*)(Whh + (size_t)row * H);
        if (s == 0) {
            tg += w0[row];
#pragma unroll 4
            for (int k = 0; k < H / 4; ++k) {
                float4 wv = whr[k], hv = hp4[k];
                th += wv.x * hv.x + wv.y * hv.y + wv.z * hv.z + wv.w * hv.w;
            }
        } else {
            tg += wb[row];
            const float4* mr = (const float4*)(Mmat + (size_t)row * H);
#pragma unroll 4
            for (int k = 0; k < H / 4; ++k) {
                float4 wv = whr[k], mv = mr[k], hv = hp4[k];
                tg += mv.x * hv.x + mv.y * hv.y + mv.z * hv.z + mv.w * hv.w;
                th += wv.x * hv.x + wv.y * hv.y + wv.z * hv.z + wv.w * hv.w;
            }
        }
        gi[q] = tg; gh[q] = th;
    }
    float r = 1.0f / (1.0f + __expf(-(gi[0] + gh[0])));
    float z = 1.0f / (1.0f + __expf(-(gi[1] + gh[1])));
    float n = tanhf(gi[2] + r * gh[2]);
    Hs[bs * H + j] = (1.0f - z) * n + z * hp[j];
}

// ---------------------------------------------------------------------------
// Kernel 4: OUT[bs][f] = Hs[bs] @ W_proj[f] + b_proj[f]
// WMMA GEMM: Mdim=64 (bs, 4 tiles), Ndim=65792 (f), K=512.
// Block 256 thr / 8 waves; each wave owns a 16-wide f strip, all 4 m-tiles.
// Hs staged whole in LDS as bf16 (64KB). W_proj streamed once (L2-resident
// after k_gemm_M touched it). grid = 65792/128 = 514.
// ---------------------------------------------------------------------------
__global__ __launch_bounds__(256)
void k_out_gemm(const float* __restrict__ Hs, const float* __restrict__ Wproj,
                const float* __restrict__ bproj, float* __restrict__ out) {
    __shared__ bf16_t Al[BS * H];       // 64 x 512 bf16 = 64KB
    const int tid  = threadIdx.x;
    const int lane = tid & 31;
    const int w    = tid >> 5;
    const int n0   = blockIdx.x * 128 + w * 16;

    {
        const float4* src = (const float4*)Hs;
        v4bf* dst = (v4bf*)Al;
#pragma unroll 4
        for (int i = tid; i < BS * H / 4; i += 256) dst[i] = cvt4(src[i]);
    }
    __syncthreads();

    v8f acc[4];
#pragma unroll
    for (int mt = 0; mt < 4; ++mt) acc[mt] = (v8f){};

    const int c  = lane & 15;
    const int k0 = (lane >> 4) << 4;                 // 0 or 16
    const float* brow = Wproj + (size_t)(n0 + c) * H;

#pragma unroll 2
    for (int kc = 0; kc < H; kc += 32) {
        // B fragment straight from global (16 contiguous f32 per lane -> bf16)
        union { v16bf v; v8bf h[2]; } ub;
        const float4* q = (const float4*)(brow + kc + k0);
        float4 f0 = q[0], f1 = q[1], f2 = q[2], f3 = q[3];
        ub.h[0] = (v8bf){(bf16_t)f0.x,(bf16_t)f0.y,(bf16_t)f0.z,(bf16_t)f0.w,
                         (bf16_t)f1.x,(bf16_t)f1.y,(bf16_t)f1.z,(bf16_t)f1.w};
        ub.h[1] = (v8bf){(bf16_t)f2.x,(bf16_t)f2.y,(bf16_t)f2.z,(bf16_t)f2.w,
                         (bf16_t)f3.x,(bf16_t)f3.y,(bf16_t)f3.z,(bf16_t)f3.w};
#pragma unroll
        for (int mt = 0; mt < 4; ++mt) {
            v16bf af = load_frag_a(&Al[(mt * 16) * H + kc], H, lane);
            acc[mt] = wmma_bf16(af, ub.v, acc[mt]);
        }
    }

    const float bp = bproj[n0 + c];
#pragma unroll
    for (int mt = 0; mt < 4; ++mt)
#pragma unroll
        for (int r = 0; r < 8; ++r) {
            int bs = mt * 16 + r + ((lane >> 4) << 3);
            out[(size_t)bs * FLAT + n0 + c] = acc[mt][r] + bp;
        }
}

// ---------------------------------------------------------------------------
extern "C" void kernel_launch(void* const* d_in, const int* in_sizes, int n_in,
                              void* d_out, int out_size, void* d_ws, size_t ws_size,
                              hipStream_t stream) {
    const float* content = (const float*)d_in[0];
    const float* style   = (const float*)d_in[1];
    const float* startt  = (const float*)d_in[2];
    const float* Wih     = (const float*)d_in[3];
    const float* Whh     = (const float*)d_in[4];
    const float* bih     = (const float*)d_in[5];
    const float* bhh     = (const float*)d_in[6];
    const float* Wproj   = (const float*)d_in[7];
    const float* bproj   = (const float*)d_in[8];
    float* out = (float*)d_out;

    // workspace layout (floats), total ~3.7 MB
    float* ws   = (float*)d_ws;
    float* Mmat = ws;                        // 1536*512
    float* w0   = Mmat + H3 * H;             // 1536
    float* wb   = w0 + H3;                   // 1536
    float* GIcs = wb + H3;                   // 64*1536
    float* Hs   = GIcs + BS * H3;            // 64*512

    k_zero<<<dim3((2 * H3 + 255) / 256), dim3(256), 0, stream>>>(w0, 2 * H3);

    k_gemm_M<<<dim3(H3 / 256, H / 128), dim3(512), 0, stream>>>(
        Wih, Wproj, startt, bproj, Mmat, w0, wb);

    k_gi_static<<<dim3(BS), dim3(256), 0, stream>>>(content, style, Wih, bih, GIcs);

    for (int s = 0; s < Ssz; ++s)
        k_gru_step<<<dim3(Bsz), dim3(512), 0, stream>>>(
            Mmat, Whh, bhh, GIcs, w0, wb, Hs, s);

    k_out_gemm<<<dim3(FLAT / 128), dim3(256), 0, stream>>>(Hs, Wproj, bproj, out);
}
